// JaccardIndex_34359738707
// MI455X (gfx1250) — 8-bit matrix operand order repaired and checked by reading, compiled, not run on hardware
//
#include <hip/hip_runtime.h>

#define NUM_CLASSES 16
#define HW (512 * 512)

typedef int v8i __attribute__((ext_vector_type(8)));

__device__ __forceinline__ unsigned onehot_eq4(unsigned x, unsigned m4) {
    unsigned v = x ^ m4;
    unsigned z = ~(v | ((v | 0x80808080u) - 0x01010101u)) & 0x80808080u;
    return z >> 7;
}

__device__ __forceinline__ unsigned shfl32(unsigned v, int srcLane) {
    return (unsigned)__builtin_amdgcn_ds_bpermute(srcLane << 2, (int)v);
}

__global__ void zero_conf_kernel(unsigned* conf) {
    conf[threadIdx.x] = 0u;
}

__global__ void __launch_bounds__(256)
jaccard_confusion_kernel(const float* __restrict__ pred,
                         const int* __restrict__ target,
                         int nPixels,
                         unsigned* __restrict__ conf_global) {
    __shared__ int conf_lds[NUM_CLASSES * NUM_CLASSES];
    for (int i = threadIdx.x; i < NUM_CLASSES * NUM_CLASSES; i += blockDim.x)
        conf_lds[i] = 0;
    __syncthreads();

    const int lane          = threadIdx.x & 31;
    const int wave          = threadIdx.x >> 5;
    const int wavesPerBlock = blockDim.x >> 5;
    const int totalWaves    = gridDim.x * wavesPerBlock;
    const int wid           = blockIdx.x * wavesPerBlock + wave;

    const int G     = nPixels >> 7;
    const int iters = (G + totalWaves - 1) / totalWaves;

    const int h  = lane >> 4;
    const int mn = lane & 15;
    const unsigned m4 = (unsigned)mn * 0x01010101u;

    v8i acc0 = {};
    v8i acc1 = {};

    for (int it = 0; it < iters; ++it) {
        const int grp = it * totalWaves + wid;
        if (grp < G) {
            const int g4 = grp * 128 + lane * 4;
            const int b  = g4 / HW;
            const int hw = g4 - b * HW;
            const float* p0 = pred + (size_t)b * (NUM_CLASSES * (size_t)HW) + hw;

            float4 best = *(const float4*)p0;
            int4 bi = {0, 0, 0, 0};
#pragma unroll
            for (int c = 1; c < NUM_CLASSES; ++c) {
                float4 v = *(const float4*)(p0 + (size_t)c * HW);
                if (v.x > best.x) { best.x = v.x; bi.x = c; }
                if (v.y > best.y) { best.y = v.y; bi.y = c; }
                if (v.z > best.z) { best.z = v.z; bi.z = c; }
                if (v.w > best.w) { best.w = v.w; bi.w = c; }
            }
            const unsigned packedP = (unsigned)bi.x | ((unsigned)bi.y << 8) |
                                     ((unsigned)bi.z << 16) | ((unsigned)bi.w << 24);

            const int4 tv = *(const int4*)(target + g4);
            const unsigned packedT = (unsigned)tv.x | ((unsigned)tv.y << 8) |
                                     ((unsigned)tv.z << 16) | ((unsigned)tv.w << 24);

            v8i a0, b0, a1, b1;
#pragma unroll
            for (int v = 0; v < 8; ++v) {
                const int kA = 4 * (v >> 1) + 2 * h + (v & 1);
                const int kB = 4 * (v >> 1) + 2 * h + (v & 1);
                a0[v] = (int)onehot_eq4(shfl32(packedP, kA),      m4);
                b0[v] = (int)onehot_eq4(shfl32(packedT, kB),      m4);
                a1[v] = (int)onehot_eq4(shfl32(packedP, 16 + kA), m4);
                b1[v] = (int)onehot_eq4(shfl32(packedT, 16 + kB), m4);
            }
            acc0 = __builtin_amdgcn_wmma_i32_16x16x64_iu8(
                false, a0, false, b0, acc0, false, false);
            acc1 = __builtin_amdgcn_wmma_i32_16x16x64_iu8(
                false, a1, false, b1, acc1, false, false);
        }
    }

#pragma unroll
    for (int r = 0; r < 8; ++r) {
        const int M = r + 8 * h;
        atomicAdd(&conf_lds[M * 16 + mn], acc0[r] + acc1[r]);
    }
    __syncthreads();
    for (int i = threadIdx.x; i < NUM_CLASSES * NUM_CLASSES; i += blockDim.x)
        atomicAdd(&conf_global[i], (unsigned)conf_lds[i]);
}

__global__ void jaccard_finalize_kernel(const unsigned* __restrict__ conf,
                                        float* __restrict__ out) {
    const int lane = threadIdx.x;
    float score = 0.0f;
    if (lane < NUM_CLASSES) {
        const int c = lane;
        unsigned inter = conf[c * 16 + c];
        unsigned rs = 0, cs = 0;
#pragma unroll
        for (int j = 0; j < NUM_CLASSES; ++j) {
            rs += conf[c * 16 + j];
            cs += conf[j * 16 + c];
        }
        const unsigned un = rs + cs - inter;
        score = (un == 0u) ? 1.0f : (float)inter / (float)un;
    }
#pragma unroll
    for (int off = 16; off > 0; off >>= 1)
        score += __shfl_down(score, off, 32);
    if (lane == 0) out[0] = score / (float)NUM_CLASSES;
}

extern "C" void kernel_launch(void* const* d_in, const int* in_sizes, int n_in,
                              void* d_out, int out_size, void* d_ws, size_t ws_size,
                              hipStream_t stream) {
    const float* pred   = (const float*)d_in[0];
    const int*   target = (const int*)d_in[1];
    float*       out    = (float*)d_out;
    unsigned*    conf   = (unsigned*)d_ws;

    const int nPixels = in_sizes[1];

    zero_conf_kernel<<<1, 256, 0, stream>>>(conf);
    jaccard_confusion_kernel<<<512, 256, 0, stream>>>(pred, target, nPixels, conf);
    jaccard_finalize_kernel<<<1, 32, 0, stream>>>(conf, out);
}
